// SelfModifyingLayer_3659312136300
// MI455X (gfx1250) — matrix operand order repaired and checked by reading, not compiled, not run on hardware
//
#include <hip/hip_runtime.h>
#include <hip/hip_bf16.h>
#include <cstdint>

// Problem constants (from reference)
#define DM 1024            // d_model = N = K
#define BB 4
#define SS 4096
#define MM (BB * SS)       // 16384 rows

typedef __bf16 bf16;
typedef bf16  v16bf __attribute__((ext_vector_type(16)));
typedef float v8f   __attribute__((ext_vector_type(8)));

// RNE float -> bf16 via bit ops
__device__ __forceinline__ bf16 f2bf(float f) {
    uint32_t u = __builtin_bit_cast(uint32_t, f);
    uint32_t r = (u + 0x7FFFu + ((u >> 16) & 1u)) >> 16;
    uint16_t s = (uint16_t)r;
    return __builtin_bit_cast(bf16, s);
}
__device__ __forceinline__ float bf2f(bf16 h) {
    uint16_t s = __builtin_bit_cast(uint16_t, h);
    uint32_t u = ((uint32_t)s) << 16;
    return __builtin_bit_cast(float, u);
}

union Frag {
    v16bf v;
    uint4 q[2];
};

// ---------------------------------------------------------------------------
// Elementwise fp32 -> (bf16 hi, bf16 lo) split: src = hi + lo + O(2^-18)
// ---------------------------------------------------------------------------
__global__ __launch_bounds__(256)
void split_kernel(const float* __restrict__ src, bf16* __restrict__ hi,
                  bf16* __restrict__ lo, int n) {
    int i = blockIdx.x * 256 + threadIdx.x;
    if (i < n) {
        float f = src[i];
        bf16 h = f2bf(f);
        hi[i] = h;
        lo[i] = f2bf(f - bf2f(h));
    }
}

// ---------------------------------------------------------------------------
// GEMM: out[m,n] = (sum_k A[m,k]*W[n,k]) + bias[n], optionally * mask[m]
// bf16x3: acc += Ah*Bh + Al*Bh + Ah*Bl  (f32 accumulate)
// Wave tile: 32(M) x 64(N) = 8 accumulators (2 M-tiles x 4 N-tiles).
// Block = 8 waves (2 waveM x 4 waveN) -> 64 x 256 block tile.
// B fragments are reused across the 2 M-tiles (halves load bytes per WMMA),
// and the two accumulator chains give the XDL pipe independent work.
// ---------------------------------------------------------------------------
__global__ __launch_bounds__(256)
void gemm_bf16x3_kernel(const bf16* __restrict__ Ah, const bf16* __restrict__ Al,
                        const bf16* __restrict__ Bh, const bf16* __restrict__ Bl,
                        const float* __restrict__ bias,
                        const float* __restrict__ mask,
                        float* __restrict__ out) {
    const int lane  = threadIdx.x & 31;
    const int wave  = threadIdx.x >> 5;
    const int waveM = wave >> 2;   // 0..1
    const int waveN = wave & 3;    // 0..3
    const int m0 = blockIdx.y * 64 + waveM * 32;   // wave covers rows m0..m0+31
    const int n0 = blockIdx.x * 256 + waveN * 64;
    const int lmod = lane & 15;
    const int hs   = lane >> 4;    // 0 for lanes 0-15, 1 for lanes 16-31

    // A fragment addressing (16-bit A 16x32 layout):
    //   lanes 0-15 : v0-3 = K 0..7,  v4-7 = K 16..23   (row M = lane)
    //   lanes 16-31: v0-3 = K 8..15, v4-7 = K 24..31   (row M = lane-16)
    const size_t arow0 = (size_t)(m0 + lmod);
    const bf16* aph0 = Ah + arow0 * DM + hs * 8;
    const bf16* apl0 = Al + arow0 * DM + hs * 8;
    const bf16* aph1 = aph0 + (size_t)16 * DM;
    const bf16* apl1 = apl0 + (size_t)16 * DM;

    v8f acc[2][4];
    #pragma unroll
    for (int j = 0; j < 4; ++j) {
        float bv = bias[n0 + 16 * j + lmod];
        #pragma unroll
        for (int r = 0; r < 8; ++r) { acc[0][j][r] = bv; acc[1][j][r] = bv; }
    }

    for (int k0 = 0; k0 < DM; k0 += 32) {
        Frag fah0, fal0, fah1, fal1;
        fah0.q[0] = *(const uint4*)(aph0 + k0);
        fah0.q[1] = *(const uint4*)(aph0 + k0 + 16);
        fal0.q[0] = *(const uint4*)(apl0 + k0);
        fal0.q[1] = *(const uint4*)(apl0 + k0 + 16);
        fah1.q[0] = *(const uint4*)(aph1 + k0);
        fah1.q[1] = *(const uint4*)(aph1 + k0 + 16);
        fal1.q[0] = *(const uint4*)(apl1 + k0);
        fal1.q[1] = *(const uint4*)(apl1 + k0 + 16);

        #pragma unroll
        for (int j = 0; j < 4; ++j) {
            // B fragment (16-bit B 32x16 layout):
            //   lanes 0-15 : K k0..k0+7, k0+16..k0+23  (N = lane)
            //   lanes 16-31: K k0+8..k0+15, k0+24..k0+31 (N = lane-16)
            const size_t n = (size_t)(n0 + 16 * j + lmod);
            const bf16* bph = Bh + n * DM + hs * 8 + k0;
            const bf16* bpl = Bl + n * DM + hs * 8 + k0;
            Frag fbh, fbl;
            fbh.q[0] = *(const uint4*)(bph);
            fbh.q[1] = *(const uint4*)(bph + 16);
            fbl.q[0] = *(const uint4*)(bpl);
            fbl.q[1] = *(const uint4*)(bpl + 16);

            acc[0][j] = __builtin_amdgcn_wmma_f32_16x16x32_bf16(
                false, fah0.v, false, fbh.v, (short)0, acc[0][j], false, false);
            acc[1][j] = __builtin_amdgcn_wmma_f32_16x16x32_bf16(
                false, fah1.v, false, fbh.v, (short)0, acc[1][j], false, false);
            acc[0][j] = __builtin_amdgcn_wmma_f32_16x16x32_bf16(
                false, fal0.v, false, fbh.v, (short)0, acc[0][j], false, false);
            acc[1][j] = __builtin_amdgcn_wmma_f32_16x16x32_bf16(
                false, fal1.v, false, fbh.v, (short)0, acc[1][j], false, false);
            acc[0][j] = __builtin_amdgcn_wmma_f32_16x16x32_bf16(
                false, fah0.v, false, fbl.v, (short)0, acc[0][j], false, false);
            acc[1][j] = __builtin_amdgcn_wmma_f32_16x16x32_bf16(
                false, fah1.v, false, fbl.v, (short)0, acc[1][j], false, false);
        }
    }

    // C/D layout: lane L (L<16): N=L, v[r] = C[m+r][N]; lanes>=16: rows m+8+r
    #pragma unroll
    for (int i = 0; i < 2; ++i) {
        const int rowbase = m0 + i * 16 + hs * 8;
        float mv[8];
        if (mask) {
            #pragma unroll
            for (int r = 0; r < 8; ++r) mv[r] = mask[rowbase + r];
        } else {
            #pragma unroll
            for (int r = 0; r < 8; ++r) mv[r] = 1.0f;
        }
        #pragma unroll
        for (int j = 0; j < 4; ++j) {
            const int n = n0 + 16 * j + lmod;
            #pragma unroll
            for (int r = 0; r < 8; ++r)
                out[(size_t)(rowbase + r) * DM + n] = acc[i][j][r] * mv[r];
        }
    }
}

// ---------------------------------------------------------------------------
// Decay scan: h_t = d*h_{t-1} + u_t along s, per (b, e) channel.
// 4096 independent recurrences; writes h as bf16 hi/lo split for GEMM2.
// ---------------------------------------------------------------------------
__global__ __launch_bounds__(256)
void scan_kernel(const float* __restrict__ u, const float* __restrict__ decay_param,
                 bf16* __restrict__ hh, bf16* __restrict__ hl) {
    const int gid = blockIdx.x * 256 + threadIdx.x;  // 0..4095
    const int b = gid >> 10;
    const int e = gid & 1023;
    const float d = 1.0f / (1.0f + __expf(-decay_param[0]));
    const size_t base = (size_t)b * SS * DM + e;
    const float* up = u + base;
    bf16* hhp = hh + base;
    bf16* hlp = hl + base;
    float h = 0.0f;
    #pragma unroll 4
    for (int s = 0; s < SS; ++s) {
        h = fmaf(d, h, up[(size_t)s * DM]);
        bf16 hb = f2bf(h);
        hhp[(size_t)s * DM] = hb;
        hlp[(size_t)s * DM] = f2bf(h - bf2f(hb));
    }
}

// ---------------------------------------------------------------------------
extern "C" void kernel_launch(void* const* d_in, const int* in_sizes, int n_in,
                              void* d_out, int out_size, void* d_ws, size_t ws_size,
                              hipStream_t stream) {
    const float* x     = (const float*)d_in[0];
    const float* mask  = (const float*)d_in[1];
    const float* W_up  = (const float*)d_in[2];
    const float* b_up  = (const float*)d_in[3];
    const float* W_f   = (const float*)d_in[4];
    const float* b_f   = (const float*)d_in[5];
    const float* decay = (const float*)d_in[6];
    float* out = (float*)d_out;

    char* ws = (char*)d_ws;
    const size_t nx = (size_t)MM * DM;       // 16,777,216 elements
    bf16*  xh    = (bf16*)ws;                 // 32 MB
    bf16*  xl    = (bf16*)(ws + nx * 2);      // 32 MB
    float* u     = (float*)(ws + nx * 4);     // 64 MB
    bf16*  wup_h = (bf16*)(ws + nx * 8);      // 2 MB each below
    bf16*  wup_l = wup_h + (size_t)DM * DM;
    bf16*  wf_h  = wup_l + (size_t)DM * DM;
    bf16*  wf_l  = wf_h + (size_t)DM * DM;
    bf16*  hh = xh;  // reuse x-split buffers after GEMM1
    bf16*  hl = xl;

    split_kernel<<<(unsigned)((nx + 255) / 256), 256, 0, stream>>>(x, xh, xl, (int)nx);
    split_kernel<<<(DM * DM + 255) / 256, 256, 0, stream>>>(W_up, wup_h, wup_l, DM * DM);
    split_kernel<<<(DM * DM + 255) / 256, 256, 0, stream>>>(W_f, wf_h, wf_l, DM * DM);

    dim3 grid(DM / 256, MM / 64);  // (4, 256)
    gemm_bf16x3_kernel<<<grid, 256, 0, stream>>>(xh, xl, wup_h, wup_l, b_up, mask, u);
    scan_kernel<<<(BB * DM) / 256, 256, 0, stream>>>(u, decay, hh, hl);
    gemm_bf16x3_kernel<<<grid, 256, 0, stream>>>(hh, hl, wf_h, wf_l, b_f, nullptr, out);
}
